// Longformer_28630251995604
// MI455X (gfx1250) — matrix operand order repaired and checked by reading, not compiled, not run on hardware
//
#include <hip/hip_runtime.h>
#include <hip/hip_fp16.h>
#include <hip/hip_bf16.h>
#include <math.h>

typedef _Float16 f16;
typedef __attribute__((ext_vector_type(16))) _Float16 v16h;
typedef __attribute__((ext_vector_type(8)))  float    v8f;
typedef __attribute__((ext_vector_type(4)))  unsigned int v4u;

union Frag { v16h h; v4u u[2]; };

static constexpr int Bb   = 4;
static constexpr int Ss   = 4096;
static constexpr int Hh   = 768;
static constexpr int NHh  = 12;
static constexpr int DHh  = 64;
static constexpr int Mtot = Bb * Ss;     // 16384
static constexpr int W1c  = 128;
static constexpr int NCH  = Ss / W1c;    // 32

__device__ inline v8f wmma_f16(const Frag& a, const Frag& b, v8f c) {
  return __builtin_amdgcn_wmma_f32_16x16x32_f16(false, a.h, false, b.h, (short)0, c, false, false);
}

// ---------------------------------------------------------------- tiled transpose + convert: dst[n][k] = (f16)src[k][n]
// grid (24, 24, L), block 256
__global__ __launch_bounds__(256) void k_cvt_t(const float* __restrict__ src, f16* __restrict__ dst) {
  __shared__ float tile[32][33];
  const int k0 = blockIdx.x * 32;
  const int n0 = blockIdx.y * 32;
  const size_t mo = (size_t)blockIdx.z * Hh * Hh;
  const int tx = threadIdx.x & 31, ty = threadIdx.x >> 5;   // 32 x 8
  const float* s = src + mo;
  f16* d = dst + mo;
  #pragma unroll
  for (int r = 0; r < 32; r += 8)
    tile[ty + r][tx] = s[(size_t)(k0 + ty + r) * Hh + n0 + tx];
  __syncthreads();
  #pragma unroll
  for (int r = 0; r < 32; r += 8)
    d[(size_t)(n0 + ty + r) * Hh + k0 + tx] = (f16)tile[tx][ty + r];
}

// ---------------------------------------------------------------- embedding build (pre-LN)
__global__ void k_embed(const float* __restrict__ x, const float* __restrict__ w_in,
                        const float* __restrict__ b_in, const float* __restrict__ pos,
                        const float* __restrict__ tok, float* __restrict__ tmp) {
  int idx = blockIdx.x * 256 + threadIdx.x;
  if (idx >= Mtot * Hh) return;
  int m = idx / Hh, j = idx - m * Hh;
  int s = m & (Ss - 1);
  tmp[idx] = x[m] * w_in[j] + b_in[j] + pos[(size_t)(s + 1) * Hh + j] + tok[j];
}

// ---------------------------------------------------------------- row LayerNorm (+optional residual), emits f32 + f16
template <bool HASBASE>
__global__ __launch_bounds__(256) void k_ln(const float* base, const float* __restrict__ add,
                                            const float* __restrict__ g, const float* __restrict__ b,
                                            float* hout, f16* __restrict__ h16) {
  __shared__ float red[256];
  const int row = blockIdx.x, t = threadIdx.x;
  const float* ar = add + (size_t)row * Hh;
  const float* br = HASBASE ? base + (size_t)row * Hh : nullptr;
  float x0 = ar[t]       + (HASBASE ? br[t]       : 0.f);
  float x1 = ar[t + 256] + (HASBASE ? br[t + 256] : 0.f);
  float x2 = ar[t + 512] + (HASBASE ? br[t + 512] : 0.f);
  red[t] = x0 + x1 + x2;
  __syncthreads();
  for (int o = 128; o > 0; o >>= 1) { if (t < o) red[t] += red[t + o]; __syncthreads(); }
  float mu = red[0] * (1.0f / (float)Hh);
  __syncthreads();
  float d0 = x0 - mu, d1 = x1 - mu, d2 = x2 - mu;
  red[t] = d0 * d0 + d1 * d1 + d2 * d2;
  __syncthreads();
  for (int o = 128; o > 0; o >>= 1) { if (t < o) red[t] += red[t + o]; __syncthreads(); }
  float r = rsqrtf(red[0] * (1.0f / (float)Hh) + 1e-12f);
  float y0 = d0 * r * g[t]       + b[t];
  float y1 = d1 * r * g[t + 256] + b[t + 256];
  float y2 = d2 * r * g[t + 512] + b[t + 512];
  float* ho = hout + (size_t)row * Hh;
  f16*  hh = h16 + (size_t)row * Hh;
  ho[t] = y0;       hh[t]       = (f16)y0;
  ho[t + 256] = y1; hh[t + 256] = (f16)y1;
  ho[t + 512] = y2; hh[t + 512] = (f16)y2;
}

// ---------------------------------------------------------------- WMMA GEMM: C[M,768] = A[M,768] * W[768,768] + bias
// WT is the PRE-TRANSPOSED weight: WT[n][k]. 256 threads, 128x128 tile, double-buffered LDS.
// MODE 0: outF = f32 row-major;  MODE 1: outH = f16 (B,NH,S,DH);  MODE 2: outH = f16 gelu row-major
template <int MODE>
__global__ __launch_bounds__(256) void k_gemm(const f16* __restrict__ A, const f16* __restrict__ WT,
                                              const float* __restrict__ bias,
                                              float* __restrict__ outF, f16* __restrict__ outH) {
  __shared__ __align__(16) f16 As[2 * 128 * 40];
  __shared__ __align__(16) f16 Bs[2 * 128 * 40];
  const int tid  = threadIdx.x;
  const int n0   = blockIdx.x * 128;
  const int m0   = blockIdx.y * 128;
  const int wave = tid >> 5, lane = tid & 31;
  const int l = lane & 15, hi = lane >> 4;
  const int koffA = hi ? 8 : 0;    // A frag: per-lane K = {0..7,16..23} / {8..15,24..31}
  const int koffB = hi ? 8 : 0;    // B frag: per-lane K = {0..7,16..23} / {8..15,24..31}
  const int mwave = (wave >> 1) * 32;
  const int nwave = (wave & 1) * 64;

  // tile staging: thread t loads half of one row for both A and B
  const int row  = tid >> 1;       // 0..127
  const int half = tid & 1;        // 0 / 1 (16 f16 each)
  const f16* gA = A  + (size_t)(m0 + row) * Hh + half * 16;
  const f16* gB = WT + (size_t)(n0 + row) * Hh + half * 16;
  const int lofs = row * 40 + half * 16;

  v4u av0, av1, bv0, bv1;
  auto gload = [&](int t) {
    const f16* pa = gA + t * 32;
    av0 = *(const v4u*)pa; av1 = *(const v4u*)(pa + 8);
    const f16* pb = gB + t * 32;
    bv0 = *(const v4u*)pb; bv1 = *(const v4u*)(pb + 8);
  };
  auto lstore = [&](int buf) {
    v4u* la = (v4u*)(As + buf * (128 * 40) + lofs);
    la[0] = av0; la[1] = av1;
    v4u* lb = (v4u*)(Bs + buf * (128 * 40) + lofs);
    lb[0] = bv0; lb[1] = bv1;
  };

  v8f acc[2][4] = {};

  gload(0); lstore(0);
  gload(1);
  __syncthreads();

  constexpr int NT = Hh / 32;   // 24 K-steps
  for (int t = 0; t < NT; t++) {
    const int cur = t & 1;
    if (t + 1 < NT) {
      lstore(cur ^ 1);
      if (t + 2 < NT) gload(t + 2);
    }
    const f16* ab = As + cur * (128 * 40);
    const f16* bb = Bs + cur * (128 * 40);
    Frag afr[2];
    #pragma unroll
    for (int mt = 0; mt < 2; mt++) {
      const f16* ar = ab + (mwave + mt * 16 + l) * 40;
      afr[mt].u[0] = *(const v4u*)(ar + koffA);
      afr[mt].u[1] = *(const v4u*)(ar + koffA + 16);
    }
    #pragma unroll
    for (int nt = 0; nt < 4; nt++) {
      Frag bfr;
      const f16* br = bb + (nwave + nt * 16 + l) * 40;
      bfr.u[0] = *(const v4u*)(br + koffB);
      bfr.u[1] = *(const v4u*)(br + koffB + 16);
      acc[0][nt] = wmma_f16(afr[0], bfr, acc[0][nt]);
      acc[1][nt] = wmma_f16(afr[1], bfr, acc[1][nt]);
    }
    __syncthreads();
  }

  #pragma unroll
  for (int mt = 0; mt < 2; mt++) {
    #pragma unroll
    for (int nt = 0; nt < 4; nt++) {
      const int n = n0 + nwave + nt * 16 + l;
      const float bn = bias[n];
      #pragma unroll
      for (int v = 0; v < 8; v++) {
        const int m = m0 + mwave + mt * 16 + (hi ? 8 + v : v);
        float val = acc[mt][nt][v] + bn;
        if (MODE == 0) {
          outF[(size_t)m * Hh + n] = val;
        } else if (MODE == 1) {
          const int bidx = m >> 12, s = m & (Ss - 1);
          const int nh = n >> 6, d = n & 63;
          outH[(((size_t)(bidx * NHh + nh)) * Ss + s) * DHh + d] = (f16)val;
        } else {
          float gg = 0.5f * val * (1.0f + erff(val * 0.70710678118654752f));
          outH[(size_t)m * Hh + n] = (f16)gg;
        }
      }
    }
  }
}

// ---------------------------------------------------------------- sliding-window attention, one (b,head,chunk) per WG
__global__ __launch_bounds__(256) void k_attn(const f16* __restrict__ q, const f16* __restrict__ k,
                                              const f16* __restrict__ v, f16* __restrict__ att) {
  extern __shared__ char dsm[];
  f16* S16   = (f16*)dsm;                                  // 128 x 384 scores -> probs (f16)
  f16* Vt    = (f16*)(dsm + 128 * 384 * 2);                // 64 x 384 transposed V
  float* red = (float*)(dsm + (128 * 384 + 64 * 384) * 2); // 256 floats

  const int c   = blockIdx.x;   // chunk
  const int bh  = blockIdx.y;   // b*NH + head
  const int tid = threadIdx.x;
  const size_t rowbase = (size_t)bh * Ss;

  // stage V^T into LDS (OOB rows -> 0)
  for (int e = tid; e < 64 * 384; e += 256) {
    int j = e >> 6, d = e & 63;
    int kpos = c * W1c - W1c + j;
    f16 val = (f16)0.f;
    if (kpos >= 0 && kpos < Ss) val = v[(rowbase + kpos) * DHh + d];
    Vt[d * 384 + j] = val;
  }

  const int wave = tid >> 5, lane = tid & 31, l = lane & 15, hi = lane >> 4;
  const int koffA = hi ? 8 : 0;
  const int koffB = hi ? 8 : 0;
  const int m0 = wave * 16;   // 8 waves x 16 query rows

  // phase 1: scores = Q K^T * 1/8  (A,B fragments straight from global, d-contiguous)
  const f16* qrow = q + (rowbase + (size_t)c * W1c + m0 + l) * DHh;
  Frag qa0, qa1;
  qa0.u[0] = *(const v4u*)(qrow + koffA);
  qa0.u[1] = *(const v4u*)(qrow + koffA + 16);
  qa1.u[0] = *(const v4u*)(qrow + 32 + koffA);
  qa1.u[1] = *(const v4u*)(qrow + 32 + koffA + 16);
  const v4u vzero = {};
  for (int nt = 0; nt < 24; nt++) {
    const int j = nt * 16 + l;
    const int kpos = c * W1c - W1c + j;
    Frag kb0, kb1;
    if (kpos >= 0 && kpos < Ss) {
      const f16* kr = k + (rowbase + kpos) * DHh;
      kb0.u[0] = *(const v4u*)(kr + koffB);
      kb0.u[1] = *(const v4u*)(kr + koffB + 16);
      kb1.u[0] = *(const v4u*)(kr + 32 + koffB);
      kb1.u[1] = *(const v4u*)(kr + 32 + koffB + 16);
    } else {
      kb0.u[0] = vzero; kb0.u[1] = vzero;
      kb1.u[0] = vzero; kb1.u[1] = vzero;
    }
    v8f acc = {};
    acc = wmma_f16(qa0, kb0, acc);
    acc = wmma_f16(qa1, kb1, acc);
    #pragma unroll
    for (int vv = 0; vv < 8; vv++) {
      int r = m0 + (hi ? 8 + vv : vv);
      S16[r * 384 + nt * 16 + l] = (f16)(acc[vv] * 0.125f);
    }
  }
  __syncthreads();

  // masked softmax: 2 threads per query row
  {
    const int row = tid >> 1, half = tid & 1;
    const int j0 = half * 192;
    int lo = row;                 { int t0 = W1c - c * W1c;             if (t0 > lo)  lo  = t0; }
    int hiv = row + 256;          { int t1 = (Ss - c * W1c) + W1c - 1;  if (t1 < hiv) hiv = t1; }
    float mx = -3.0e38f;
    for (int j = j0; j < j0 + 192; j++)
      if (j >= lo && j <= hiv) mx = fmaxf(mx, (float)S16[row * 384 + j]);
    red[tid] = mx;
    __syncthreads();
    float rmx = fmaxf(red[row * 2], red[row * 2 + 1]);
    __syncthreads();
    float sum = 0.f;
    for (int j = j0; j < j0 + 192; j++)
      if (j >= lo && j <= hiv) sum += __expf((float)S16[row * 384 + j] - rmx);
    red[tid] = sum;
    __syncthreads();
    float inv = 1.0f / (red[row * 2] + red[row * 2 + 1]);
    for (int j = j0; j < j0 + 192; j++) {
      float p = 0.f;
      if (j >= lo && j <= hiv) p = __expf((float)S16[row * 384 + j] - rmx) * inv;
      S16[row * 384 + j] = (f16)p;
    }
  }
  __syncthreads();

  // phase 2: out = P @ V
  v8f oacc[4] = {};
  for (int kt = 0; kt < 12; kt++) {
    const int jb = kt * 32;
    Frag pa;
    const f16* pr = S16 + (m0 + l) * 384 + jb;
    pa.u[0] = *(const v4u*)(pr + koffA);
    pa.u[1] = *(const v4u*)(pr + koffA + 16);
    #pragma unroll
    for (int nt = 0; nt < 4; nt++) {
      Frag vb;
      const f16* vr = Vt + (nt * 16 + l) * 384 + jb;
      vb.u[0] = *(const v4u*)(vr + koffB);
      vb.u[1] = *(const v4u*)(vr + koffB + 16);
      oacc[nt] = wmma_f16(pa, vb, oacc[nt]);
    }
  }
  const int bidx = bh / NHh, nh = bh % NHh;
  #pragma unroll
  for (int nt = 0; nt < 4; nt++) {
    const int d = nt * 16 + l;
    #pragma unroll
    for (int vv = 0; vv < 8; vv++) {
      const int mrow = m0 + (hi ? 8 + vv : vv);
      att[((size_t)bidx * Ss + (size_t)c * W1c + mrow) * Hh + nh * DHh + d] = (f16)oacc[nt][vv];
    }
  }
}

// ---------------------------------------------------------------- two-stage mean over S, then tiny head
// stage 1: grid (Bb, 32), block 256: partial column sums over 128 rows each
__global__ __launch_bounds__(256) void k_mean1(const float* __restrict__ h, float* __restrict__ part) {
  const int b = blockIdx.x, chunk = blockIdx.y, t = threadIdx.x;
  const float* hp = h + ((size_t)b * Ss + (size_t)chunk * 128) * Hh;
  #pragma unroll
  for (int i = 0; i < 3; i++) {
    const int j = t + i * 256;
    float s = 0.f;
    for (int r = 0; r < 128; r++) s += hp[(size_t)r * Hh + j];
    part[((size_t)b * 32 + chunk) * Hh + j] = s;
  }
}

// stage 2: grid (Bb), block 768: fold 32 partials
__global__ void k_mean2(const float* __restrict__ part, float* __restrict__ mb) {
  const int b = blockIdx.x, j = threadIdx.x;
  float s = 0.f;
  #pragma unroll
  for (int cchunk = 0; cchunk < 32; cchunk++) s += part[((size_t)b * 32 + cchunk) * Hh + j];
  mb[b * Hh + j] = s * (1.0f / (float)Ss);
}

__global__ void k_head(const float* __restrict__ mb, const float* __restrict__ w_out,
                       const float* __restrict__ b_out, float* __restrict__ out) {
  int t = threadIdx.x;
  if (t >= Bb * 10) return;
  int b = t / 10, o = t - b * 10;
  float s = b_out[o];
  for (int j = 0; j < Hh; j++) s += mb[b * Hh + j] * w_out[j * 10 + o];
  out[t] = s;
}

// ---------------------------------------------------------------- host
extern "C" void kernel_launch(void* const* d_in, const int* in_sizes, int n_in,
                              void* d_out, int out_size, void* d_ws, size_t ws_size,
                              hipStream_t stream) {
  (void)in_sizes; (void)n_in; (void)out_size; (void)ws_size;
  const float* x       = (const float*)d_in[0];
  const float* w_in    = (const float*)d_in[1];
  const float* b_in    = (const float*)d_in[2];
  const float* pos_emb = (const float*)d_in[3];
  const float* tok_emb = (const float*)d_in[4];
  const float* ln_eg   = (const float*)d_in[5];
  const float* ln_eb   = (const float*)d_in[6];
  const float* Wq = (const float*)d_in[7];   const float* bq = (const float*)d_in[8];
  const float* Wk = (const float*)d_in[9];   const float* bk = (const float*)d_in[10];
  const float* Wv = (const float*)d_in[11];  const float* bv = (const float*)d_in[12];
  const float* Wo = (const float*)d_in[13];  const float* bo = (const float*)d_in[14];
  const float* ln1g = (const float*)d_in[15]; const float* ln1b = (const float*)d_in[16];
  const float* Wi = (const float*)d_in[17];  const float* bi = (const float*)d_in[18];
  const float* Wf = (const float*)d_in[19];  const float* bf = (const float*)d_in[20];
  const float* ln2g = (const float*)d_in[21]; const float* ln2b = (const float*)d_in[22];
  const float* w_out = (const float*)d_in[23]; const float* b_out = (const float*)d_in[24];
  float* out = (float*)d_out;

  char* ws = (char*)d_ws;
  size_t off = 0;
  auto alloc = [&](size_t bytes) { char* p = ws + off; off += (bytes + 255) & ~(size_t)255; return p; };
  const size_t MHf = (size_t)Mtot * Hh;
  float* h    = (float*)alloc(MHf * 4);
  float* tmp  = (float*)alloc(MHf * 4);
  f16* h16    = (f16*)alloc(MHf * 2);
  f16* q16    = (f16*)alloc(MHf * 2);
  f16* k16    = (f16*)alloc(MHf * 2);
  f16* v16    = (f16*)alloc(MHf * 2);
  f16* att16  = (f16*)alloc(MHf * 2);
  f16* fi16   = (f16*)alloc(MHf * 2);
  const size_t wfam = (size_t)4 * Hh * Hh;  // per family: L * H * H elements (transposed f16)
  f16* w16q = (f16*)alloc(wfam * 2);
  f16* w16k = (f16*)alloc(wfam * 2);
  f16* w16v = (f16*)alloc(wfam * 2);
  f16* w16o = (f16*)alloc(wfam * 2);
  f16* w16i = (f16*)alloc(wfam * 2);
  f16* w16f = (f16*)alloc(wfam * 2);
  float* part  = (float*)alloc((size_t)Bb * 32 * Hh * 4);
  float* meanb = (float*)alloc((size_t)Bb * Hh * 4);

  // 1) weight conversion to transposed f16 (WT[n][k])
  const dim3 tgrid(Hh / 32, Hh / 32, 4);
  k_cvt_t<<<tgrid, 256, 0, stream>>>(Wq, w16q);
  k_cvt_t<<<tgrid, 256, 0, stream>>>(Wk, w16k);
  k_cvt_t<<<tgrid, 256, 0, stream>>>(Wv, w16v);
  k_cvt_t<<<tgrid, 256, 0, stream>>>(Wo, w16o);
  k_cvt_t<<<tgrid, 256, 0, stream>>>(Wi, w16i);
  k_cvt_t<<<tgrid, 256, 0, stream>>>(Wf, w16f);

  // 2) embedding + first LN
  k_embed<<<(int)((MHf + 255) / 256), 256, 0, stream>>>(x, w_in, b_in, pos_emb, tok_emb, tmp);
  k_ln<false><<<Mtot, 256, 0, stream>>>(nullptr, tmp, ln_eg, ln_eb, h, h16);

  const dim3 ggrid(Hh / 128, Mtot / 128);          // (6, 128)
  const size_t attshm = (size_t)(128 * 384 + 64 * 384) * 2 + 256 * 4;

  for (int l = 0; l < 4; l++) {
    const size_t wof = (size_t)l * Hh * Hh;
    const size_t bof = (size_t)l * Hh;
    // QKV projections -> f16 (B,NH,S,DH)
    k_gemm<1><<<ggrid, 256, 0, stream>>>(h16, w16q + wof, bq + bof, nullptr, q16);
    k_gemm<1><<<ggrid, 256, 0, stream>>>(h16, w16k + wof, bk + bof, nullptr, k16);
    k_gemm<1><<<ggrid, 256, 0, stream>>>(h16, w16v + wof, bv + bof, nullptr, v16);
    // sliding window attention
    k_attn<<<dim3(NCH, Bb * NHh), 256, attshm, stream>>>(q16, k16, v16, att16);
    // output projection + residual LN
    k_gemm<0><<<ggrid, 256, 0, stream>>>(att16, w16o + wof, bo + bof, tmp, nullptr);
    k_ln<true><<<Mtot, 256, 0, stream>>>(h, tmp, ln1g + bof, ln1b + bof, h, h16);
    // FFN: gelu(h Wi + bi) Wf + bf, residual LN
    k_gemm<2><<<ggrid, 256, 0, stream>>>(h16, w16i + wof, bi + bof, nullptr, fi16);
    k_gemm<0><<<ggrid, 256, 0, stream>>>(fi16, w16f + wof, bf + bof, tmp, nullptr);
    k_ln<true><<<Mtot, 256, 0, stream>>>(h, tmp, ln2g + bof, ln2b + bof, h, h16);
  }

  // 3) two-stage mean over sequence + output head
  k_mean1<<<dim3(Bb, 32), 256, 0, stream>>>(h, part);
  k_mean2<<<Bb, Hh, 0, stream>>>(part, meanb);
  k_head<<<1, 64, 0, stream>>>(meanb, w_out, b_out, out);
}
